// PFNet5_15814069584533
// MI455X (gfx1250) — 16-bit and 8-bit matrix operand order repaired and checked by reading, compiled, not run on hardware
//
#include <hip/hip_runtime.h>
#include <math.h>

#define Bc   16
#define Nc   1024
#define INC  3
#define HIDc 32
#define OUTC 4
#define K1c  205
#define K1Pc 208
#define K2c  41
#define K2Pc 48
#define BN_EPSc 1e-5f

typedef __attribute__((ext_vector_type(16))) _Float16 v16h;
typedef __attribute__((ext_vector_type(8)))  float    v8f;
typedef __attribute__((ext_vector_type(16))) int      v16i;

#define FP8_ONE 0x38

__device__ __forceinline__ void byte4_to_h2(unsigned x, unsigned& d0, unsigned& d1) {
  d0 = ((x & 0x000000FFu) ? 0x00003C00u : 0u) | ((x & 0x0000FF00u) ? 0x3C000000u : 0u);
  d1 = ((x & 0x00FF0000u) ? 0x00003C00u : 0u) | ((x & 0xFF000000u) ? 0x3C000000u : 0u);
}

__global__ void k_to_fp8(const float* __restrict__ adj, unsigned char* __restrict__ a8,
                         long total) {
  long i = (long)blockIdx.x * blockDim.x + threadIdx.x;
  if (i < total) a8[i] = (adj[i] != 0.0f) ? (unsigned char)FP8_ONE : (unsigned char)0;
}

__global__ void k_deg_full(const float* __restrict__ adj, float* __restrict__ dis) {
  __shared__ float red[128];
  int row = blockIdx.x;
  const float* a = adj + (size_t)row * Nc;
  float s = 0.f;
  for (int j = threadIdx.x; j < Nc; j += 128) s += a[j];
  red[threadIdx.x] = s; __syncthreads();
  for (int st = 64; st > 0; st >>= 1) {
    if (threadIdx.x < st) red[threadIdx.x] += red[threadIdx.x + st];
    __syncthreads();
  }
  if (threadIdx.x == 0) dis[row] = rsqrtf(red[0] + 2.0f);
}

__global__ void k_prep(const float* __restrict__ xin, int indim,
                       const float* __restrict__ W, const float* __restrict__ dis,
                       float* __restrict__ s32, _Float16* __restrict__ s16T, int NN) {
  int idx = blockIdx.x * blockDim.x + threadIdx.x;
  int total = Bc * NN * HIDc;
  if (idx >= total) return;
  int f    = idx & (HIDc - 1);
  int node = (idx / HIDc) % NN;
  int b    = idx / (HIDc * NN);
  const float* xr = xin + (size_t)(b * NN + node) * indim;
  float a = 0.f;
  for (int c = 0; c < indim; ++c) a += xr[c] * W[c * HIDc + f];
  float s = dis[b * NN + node] * a;
  s32[(size_t)(b * NN + node) * HIDc + f] = s;
  if (s16T) s16T[((size_t)(b * HIDc + f)) * Nc + node] = (_Float16)s;
}

__global__ __launch_bounds__(32)
void k_spmm_wmma(const unsigned char* __restrict__ A8, const float* __restrict__ s32,
                 const _Float16* __restrict__ s16T, const float* __restrict__ dis,
                 const float* __restrict__ bias, float* __restrict__ out, int relu) {
  int b    = blockIdx.x >> 6;
  int mt   = blockIdx.x & 63;
  int lane = threadIdx.x;
  int half = lane >> 4;
  int l15  = lane & 15;
  const unsigned char* arow = A8 + ((size_t)(b * Nc + mt * 16 + l15)) * Nc;
  const _Float16* b0row = s16T + ((size_t)(b * HIDc + l15)) * Nc;
  const _Float16* b1row = s16T + ((size_t)(b * HIDc + 16 + l15)) * Nc;
  int koffA = half * 16;
  int koffB = half * 16;
  v8f acc0 = {};
  v8f acc1 = {};
  for (int kt = 0; kt < Nc / 32; ++kt) {
    int kbase = kt * 32;
    const unsigned* pa = (const unsigned*)(arow + kbase + koffA);
    unsigned lo0 = pa[0], lo1 = pa[1];
    const unsigned* pb = (const unsigned*)(arow + kbase + 8 + koffA);
    unsigned hi0 = pb[0], hi1 = pb[1];
    union { unsigned u[8]; v16h h; } af;
    byte4_to_h2(lo0, af.u[0], af.u[1]);
    byte4_to_h2(lo1, af.u[2], af.u[3]);
    byte4_to_h2(hi0, af.u[4], af.u[5]);
    byte4_to_h2(hi1, af.u[6], af.u[7]);
    union { unsigned u[8]; v16h h; } bf0, bf1;
#pragma unroll
    for (int d = 0; d < 8; ++d) {
      int k0 = kbase + koffB + d * 2;
      bf0.u[d] = *(const unsigned*)(b0row + k0);
      bf1.u[d] = *(const unsigned*)(b1row + k0);
    }
    acc0 = __builtin_amdgcn_wmma_f32_16x16x32_f16(false, af.h, false, bf0.h,
                                                  (short)0, acc0, false, false);
    acc1 = __builtin_amdgcn_wmma_f32_16x16x32_f16(false, af.h, false, bf1.h,
                                                  (short)0, acc1, false, false);
  }
#pragma unroll
  for (int v = 0; v < 8; ++v) {
    int i = mt * 16 + v + half * 8;
    size_t base = (size_t)(b * Nc + i) * HIDc;
    float di = dis[b * Nc + i];
    int f0 = l15, f1 = 16 + l15;
    float o0 = di * (acc0[v] + 2.0f * s32[base + f0]) + bias[f0];
    float o1 = di * (acc1[v] + 2.0f * s32[base + f1]) + bias[f1];
    if (relu) { o0 = fmaxf(o0, 0.f); o1 = fmaxf(o1, 0.f); }
    out[base + f0] = o0;
    out[base + f1] = o1;
  }
}

template <int SORTN>
__global__ __launch_bounds__(SORTN)
void k_topk(const float* __restrict__ h, int nodestride, int nvalid,
            const float* __restrict__ p, int* __restrict__ perm,
            float* __restrict__ vals, int KP, int Kvalid) {
  __shared__ float sval[SORTN];
  __shared__ int   sidx[SORTN];
  int b = blockIdx.x, tid = threadIdx.x;
  float nrm = 0.f;
  for (int c = 0; c < HIDc; ++c) { float w = p[c]; nrm += w * w; }
  nrm = sqrtf(nrm);
  float sc = -3.0e38f;
  if (tid < nvalid) {
    const float* hr = h + (size_t)(b * nodestride + tid) * HIDc;
    float d = 0.f;
    for (int c = 0; c < HIDc; ++c) d += hr[c] * p[c];
    sc = tanhf(d / nrm);
  }
  sval[tid] = sc; sidx[tid] = tid;
  __syncthreads();
  for (int k = 2; k <= SORTN; k <<= 1) {
    for (int j = k >> 1; j > 0; j >>= 1) {
      int ixj = tid ^ j;
      if (ixj > tid) {
        bool asc = ((tid & k) == 0);
        float va = sval[tid], vb = sval[ixj];
        int   ia = sidx[tid], ib = sidx[ixj];
        bool before = (va > vb) || (va == vb && ia < ib);
        if (asc ? !before : before) {
          sval[tid] = vb; sidx[tid] = ib;
          sval[ixj] = va; sidx[ixj] = ia;
        }
      }
      __syncthreads();
    }
  }
  if (tid < KP) {
    perm[b * KP + tid] = (tid < Kvalid) ? sidx[tid] : -1;
    vals[b * KP + tid] = (tid < Kvalid) ? sval[tid] : 0.f;
  }
}

__global__ void k_pool(const float* __restrict__ h, const int* __restrict__ perm,
                       const float* __restrict__ vals, float* __restrict__ hp,
                       int KP, int srcstride) {
  int idx = blockIdx.x * blockDim.x + threadIdx.x;
  int total = Bc * KP * HIDc;
  if (idx >= total) return;
  int f = idx & (HIDc - 1);
  int pp = (idx / HIDc) % KP;
  int b  = idx / (HIDc * KP);
  int u = perm[b * KP + pp];
  float v = 0.f;
  if (u >= 0) v = h[(size_t)(b * srcstride + u) * HIDc + f] * vals[b * KP + pp];
  hp[(size_t)(b * KP + pp) * HIDc + f] = v;
}

__global__ void k_gather8(const unsigned char* __restrict__ A8,
                          const int* __restrict__ perm, unsigned char* __restrict__ G8) {
  int idx = blockIdx.x * blockDim.x + threadIdx.x;
  int total = Bc * K1Pc * (Nc / 4);
  if (idx >= total) return;
  int c  = idx % (Nc / 4);
  int pp = (idx / (Nc / 4)) % K1Pc;
  int b  = idx / ((Nc / 4) * K1Pc);
  int u = perm[b * K1Pc + pp];
  unsigned v = 0u;
  if (u >= 0) v = ((const unsigned*)A8)[(size_t)(b * Nc + u) * (Nc / 4) + c];
  ((unsigned*)G8)[(size_t)idx] = v;
}

__global__ __launch_bounds__(32)
void k_augment_wmma(const unsigned char* __restrict__ G8, const float* __restrict__ adj,
                    const int* __restrict__ perm, float* __restrict__ A1) {
  const int T = K1Pc / 16;
  int b  = blockIdx.x / (T * T);
  int r  = blockIdx.x % (T * T);
  int pt = r / T, qt = r % T;
  int lane = threadIdx.x;
  int half = lane >> 4;
  int l15  = lane & 15;
  const unsigned char* prow = G8 + ((size_t)(b * K1Pc + pt * 16 + l15)) * Nc;
  const unsigned char* qrow = G8 + ((size_t)(b * K1Pc + qt * 16 + l15)) * Nc;
  v8f acc = {};
  for (int kt = 0; kt < Nc / 128; ++kt) {
    int kbase = kt * 128;
    union { int i[16]; v16i v; } af, bf;
#pragma unroll
    for (int d = 0; d < 16; ++d) {
      int ka = kbase + ((d >= 8) ? 64 : 0) + ((d & 7) >> 1) * 16 + (d & 1) * 4 + half * 8;
      af.i[d] = *(const int*)(prow + ka);
      int kb = kbase + ((d >= 8) ? 64 : 0) + ((d & 7) >> 1) * 16 + (d & 1) * 4 + half * 8;
      bf.i[d] = *(const int*)(qrow + kb);
    }
    acc = __builtin_amdgcn_wmma_f32_16x16x128_fp8_fp8(af.v, bf.v, (short)0, acc,
                                                      false, false);
  }
  int q = qt * 16 + l15;
  int w = perm[b * K1Pc + q];
#pragma unroll
  for (int v = 0; v < 8; ++v) {
    int p = pt * 16 + v + half * 8;
    int u = perm[b * K1Pc + p];
    float val = 0.f;
    if (p != q && u >= 0 && w >= 0)
      val = acc[v] + 2.0f * adj[((size_t)(b * Nc + u)) * Nc + w];
    A1[((size_t)(b * K1Pc + p)) * K1Pc + q] = val;
  }
}

__global__ void k_deg_small(const float* __restrict__ A, float* __restrict__ dis, int KP) {
  int idx = blockIdx.x * blockDim.x + threadIdx.x;
  if (idx >= Bc * KP) return;
  const float* r = A + (size_t)idx * KP;
  float s = 0.f;
  for (int j = 0; j < KP; ++j) s += r[j];
  dis[idx] = rsqrtf(s + 2.0f);
}

__global__ void k_spmm_small(const float* __restrict__ A, const float* __restrict__ s32,
                             const float* __restrict__ dis, const float* __restrict__ bias,
                             float* __restrict__ out, int relu, int KP) {
  int idx = blockIdx.x * blockDim.x + threadIdx.x;
  int total = Bc * KP * HIDc;
  if (idx >= total) return;
  int f = idx & (HIDc - 1);
  int i = (idx / HIDc) % KP;
  int b = idx / (HIDc * KP);
  const float* ar = A + (size_t)(b * KP + i) * KP;
  const float* sB = s32 + (size_t)b * KP * HIDc;
  float acc = 0.f;
  for (int j = 0; j < KP; ++j) acc += ar[j] * sB[j * HIDc + f];
  float di = dis[b * KP + i];
  float o = di * acc + 2.0f * di * s32[(size_t)(b * KP + i) * HIDc + f] + bias[f];
  if (relu) o = fmaxf(o, 0.f);
  out[(size_t)(b * KP + i) * HIDc + f] = o;
}

__global__ void k_augment2(const float* __restrict__ A1, const int* __restrict__ perm2,
                           float* __restrict__ A2) {
  int idx = blockIdx.x * blockDim.x + threadIdx.x;
  int total = Bc * K2Pc * K2Pc;
  if (idx >= total) return;
  int q  = idx % K2Pc;
  int pp = (idx / K2Pc) % K2Pc;
  int b  = idx / (K2Pc * K2Pc);
  int u = perm2[b * K2Pc + pp], v = perm2[b * K2Pc + q];
  float val = 0.f;
  if (pp != q && u >= 0 && v >= 0) {
    const float* ru = A1 + (size_t)(b * K1Pc + u) * K1Pc;
    const float* rv = A1 + (size_t)(b * K1Pc + v) * K1Pc;
    float acc = 0.f;
    for (int k = 0; k < K1Pc; ++k) {
      float bu = ru[k] + (u == k ? 1.f : 0.f);
      float bv = rv[k] + (v == k ? 1.f : 0.f);
      acc += bu * bv;
    }
    val = acc;
  }
  A2[(size_t)(b * K2Pc + pp) * K2Pc + q] = val;
}

__global__ void k_copy(float* __restrict__ dst, const float* __restrict__ src, long n) {
  long i = (long)blockIdx.x * blockDim.x + threadIdx.x;
  if (i < n) dst[i] = src[i];
}

__global__ void k_scatter_add(float* __restrict__ dst, const float* __restrict__ src,
                              const int* __restrict__ perm, int KP, int dststride) {
  int idx = blockIdx.x * blockDim.x + threadIdx.x;
  int total = Bc * KP * HIDc;
  if (idx >= total) return;
  int f  = idx & (HIDc - 1);
  int pp = (idx / HIDc) % KP;
  int b  = idx / (HIDc * KP);
  int u = perm[b * KP + pp];
  if (u >= 0)
    dst[(size_t)(b * dststride + u) * HIDc + f] += src[(size_t)(b * KP + pp) * HIDc + f];
}

__global__ __launch_bounds__(256)
void k_bn_partial(const float* __restrict__ h, float* __restrict__ part) {
  __shared__ float ls[256], lq[256];
  int tid = threadIdx.x;
  int f = tid & 31, g = tid >> 5;
  const int rowsPerBlock = (Bc * Nc) / 64;
  int r0 = blockIdx.x * rowsPerBlock;
  float s = 0.f, q = 0.f;
  for (int r = g; r < rowsPerBlock; r += 8) {
    float v = h[(size_t)(r0 + r) * HIDc + f];
    s += v; q += v * v;
  }
  ls[tid] = s; lq[tid] = q; __syncthreads();
  if (tid < 32) {
    float ss = 0.f, qq = 0.f;
    for (int gg = 0; gg < 8; ++gg) { ss += ls[tid + 32 * gg]; qq += lq[tid + 32 * gg]; }
    part[blockIdx.x * 64 + tid]      = ss;
    part[blockIdx.x * 64 + 32 + tid] = qq;
  }
}

__global__ void k_bn_reduce(const float* __restrict__ part, float* __restrict__ stats,
                            int nblk) {
  int tid = threadIdx.x;
  if (tid < 64) {
    float s = 0.f;
    for (int i = 0; i < nblk; ++i) s += part[i * 64 + tid];
    stats[tid] = s;
  }
}

__global__ __launch_bounds__(256)
void k_mlp(const float* __restrict__ h, const float* __restrict__ stats,
           const float* __restrict__ gamma, const float* __restrict__ beta,
           const float* __restrict__ W1, const float* __restrict__ b1,
           const float* __restrict__ W2, const float* __restrict__ b2,
           const float* __restrict__ W3, const float* __restrict__ b3,
           float* __restrict__ out) {
  __shared__ float sW1[HIDc * HIDc], sW2[HIDc * HIDc], sW3[HIDc * OUTC];
  __shared__ float sb1[HIDc], sb2[HIDc], sb3[OUTC];
  __shared__ float sm[HIDc], siv[HIDc], sg[HIDc], sbt[HIDc];
  int tid = threadIdx.x;
  for (int i = tid; i < HIDc * HIDc; i += blockDim.x) { sW1[i] = W1[i]; sW2[i] = W2[i]; }
  for (int i = tid; i < HIDc * OUTC; i += blockDim.x) sW3[i] = W3[i];
  if (tid < HIDc) {
    sb1[tid] = b1[tid]; sb2[tid] = b2[tid]; sg[tid] = gamma[tid]; sbt[tid] = beta[tid];
    const float inv = 1.0f / (float)(Bc * Nc);
    float mean = stats[tid] * inv;
    float var  = stats[32 + tid] * inv - mean * mean;
    sm[tid] = mean; siv[tid] = rsqrtf(var + BN_EPSc);
  }
  if (tid < OUTC) sb3[tid] = b3[tid];
  __syncthreads();
  int node = blockIdx.x * blockDim.x + tid;
  if (node >= Bc * Nc) return;
  const float* hr = h + (size_t)node * HIDc;
  float hn[HIDc];
  for (int i = 0; i < HIDc; ++i) hn[i] = (hr[i] - sm[i]) * siv[i] * sg[i] + sbt[i];
  float r1[HIDc];
  for (int o = 0; o < HIDc; ++o) {
    float a = sb1[o];
    for (int i = 0; i < HIDc; ++i) a += hn[i] * sW1[i * HIDc + o];
    r1[o] = a > 0.f ? a : 0.01f * a;
  }
  float r2[HIDc];
  for (int o = 0; o < HIDc; ++o) {
    float a = sb2[o];
    for (int i = 0; i < HIDc; ++i) a += r1[i] * sW2[i * HIDc + o];
    r2[o] = a > 0.f ? a : 0.01f * a;
  }
  float r3[OUTC];
  for (int o = 0; o < OUTC; ++o) {
    float a = sb3[o];
    for (int i = 0; i < HIDc; ++i) a += r2[i] * sW3[i * OUTC + o];
    r3[o] = a;
  }
  r3[0] = 1.0f / (1.0f + expf(-r3[0]));
  for (int o = 0; o < OUTC; ++o) out[(size_t)node * OUTC + o] = r3[o];
}

extern "C" void kernel_launch(void* const* d_in, const int* in_sizes, int n_in,
                              void* d_out, int out_size, void* d_ws, size_t ws_size,
                              hipStream_t stream) {
  const float* x    = (const float*)d_in[0];
  const float* adj  = (const float*)d_in[1];
  const float* Wd0  = (const float*)d_in[2];
  const float* bd0  = (const float*)d_in[3];
  const float* Wd1  = (const float*)d_in[4];
  const float* bd1  = (const float*)d_in[5];
  const float* Wd2  = (const float*)d_in[6];
  const float* bd2  = (const float*)d_in[7];
  const float* Wu0  = (const float*)d_in[8];
  const float* bu0  = (const float*)d_in[9];
  const float* Wu1  = (const float*)d_in[10];
  const float* bu1  = (const float*)d_in[11];
  const float* p1   = (const float*)d_in[12];
  const float* p2   = (const float*)d_in[13];
  const float* gmm  = (const float*)d_in[14];
  const float* bet  = (const float*)d_in[15];
  const float* W1   = (const float*)d_in[16];
  const float* b1   = (const float*)d_in[17];
  const float* W2   = (const float*)d_in[18];
  const float* b2   = (const float*)d_in[19];
  const float* W3   = (const float*)d_in[20];
  const float* b3   = (const float*)d_in[21];
  float* out = (float*)d_out;
  (void)in_sizes; (void)n_in; (void)out_size; (void)ws_size;

  char* w = (char*)d_ws;
  size_t off = 0;
  auto alloc = [&](size_t bytes) -> void* {
    void* p = w + off;
    off += (bytes + 255) & ~(size_t)255;
    return p;
  };
  unsigned char* A8   = (unsigned char*)alloc((size_t)Bc * Nc * Nc);
  float*     dis0 = (float*)    alloc((size_t)Bc * Nc * 4);
  float*     s32  = (float*)    alloc((size_t)Bc * Nc * HIDc * 4);
  _Float16*  s16T = (_Float16*) alloc((size_t)Bc * HIDc * Nc * 2);
  float*     h0   = (float*)    alloc((size_t)Bc * Nc * HIDc * 4);
  int*       perm1= (int*)      alloc((size_t)Bc * K1Pc * 4);
  float*     vals1= (float*)    alloc((size_t)Bc * K1Pc * 4);
  float*     hp1  = (float*)    alloc((size_t)Bc * K1Pc * HIDc * 4);
  unsigned char* G8 = (unsigned char*)alloc((size_t)Bc * K1Pc * Nc);
  float*     A1   = (float*)    alloc((size_t)Bc * K1Pc * K1Pc * 4);
  float*     dis1 = (float*)    alloc((size_t)Bc * K1Pc * 4);
  float*     sS   = (float*)    alloc((size_t)Bc * K1Pc * HIDc * 4);
  float*     h1   = (float*)    alloc((size_t)Bc * K1Pc * HIDc * 4);
  int*       perm2= (int*)      alloc((size_t)Bc * K2Pc * 4);
  float*     vals2= (float*)    alloc((size_t)Bc * K2Pc * 4);
  float*     hp2  = (float*)    alloc((size_t)Bc * K2Pc * HIDc * 4);
  float*     A2   = (float*)    alloc((size_t)Bc * K2Pc * K2Pc * 4);
  float*     dis2 = (float*)    alloc((size_t)Bc * K2Pc * 4);
  float*     sS2  = (float*)    alloc((size_t)Bc * K2Pc * HIDc * 4);
  float*     h2   = (float*)    alloc((size_t)Bc * K2Pc * HIDc * 4);
  float*     hu   = (float*)    alloc((size_t)Bc * K1Pc * HIDc * 4);
  float*     g1   = (float*)    alloc((size_t)Bc * K1Pc * HIDc * 4);
  float*     hN   = (float*)    alloc((size_t)Bc * Nc * HIDc * 4);
  float*     hfin = (float*)    alloc((size_t)Bc * Nc * HIDc * 4);
  float*     part = (float*)    alloc((size_t)64 * 64 * 4);
  float*     stats= (float*)    alloc((size_t)64 * 4);

  const int TB = 256;
  auto grid = [](long n, int tb) { return (unsigned)((n + tb - 1) / tb); };

  long nAdj = (long)Bc * Nc * Nc;
  k_to_fp8<<<grid(nAdj, TB), TB, 0, stream>>>(adj, A8, nAdj);
  k_deg_full<<<Bc * Nc, 128, 0, stream>>>(adj, dis0);

  k_prep<<<grid((long)Bc * Nc * HIDc, TB), TB, 0, stream>>>(x, INC, Wd0, dis0, s32, s16T, Nc);
  k_spmm_wmma<<<Bc * (Nc / 16), 32, 0, stream>>>(A8, s32, s16T, dis0, bd0, h0, 1);

  k_topk<1024><<<Bc, 1024, 0, stream>>>(h0, Nc, Nc, p1, perm1, vals1, K1Pc, K1c);
  k_pool<<<grid((long)Bc * K1Pc * HIDc, TB), TB, 0, stream>>>(h0, perm1, vals1, hp1, K1Pc, Nc);
  k_gather8<<<grid((long)Bc * K1Pc * (Nc / 4), TB), TB, 0, stream>>>(A8, perm1, G8);
  k_augment_wmma<<<Bc * (K1Pc / 16) * (K1Pc / 16), 32, 0, stream>>>(G8, adj, perm1, A1);

  k_deg_small<<<grid((long)Bc * K1Pc, TB), TB, 0, stream>>>(A1, dis1, K1Pc);
  k_prep<<<grid((long)Bc * K1Pc * HIDc, TB), TB, 0, stream>>>(hp1, HIDc, Wd1, dis1, sS, (
      _Float16*)nullptr, K1Pc);
  k_spmm_small<<<grid((long)Bc * K1Pc * HIDc, TB), TB, 0, stream>>>(A1, sS, dis1, bd1, h1, 1, K1Pc);

  k_topk<256><<<Bc, 256, 0, stream>>>(h1, K1Pc, K1c, p2, perm2, vals2, K2Pc, K2c);
  k_pool<<<grid((long)Bc * K2Pc * HIDc, TB), TB, 0, stream>>>(h1, perm2, vals2, hp2, K2Pc, K1Pc);
  k_augment2<<<grid((long)Bc * K2Pc * K2Pc, TB), TB, 0, stream>>>(A1, perm2, A2);

  k_deg_small<<<grid((long)Bc * K2Pc, TB), TB, 0, stream>>>(A2, dis2, K2Pc);
  k_prep<<<grid((long)Bc * K2Pc * HIDc, TB), TB, 0, stream>>>(hp2, HIDc, Wd2, dis2, sS2, (
      _Float16*)nullptr, K2Pc);
  k_spmm_small<<<grid((long)Bc * K2Pc * HIDc, TB), TB, 0, stream>>>(A2, sS2, dis2, bd2, h2, 1, K2Pc);

  k_copy<<<grid((long)Bc * K1Pc * HIDc, TB), TB, 0, stream>>>(hu, h1, (long)Bc * K1Pc * HIDc);
  k_scatter_add<<<grid((long)Bc * K2Pc * HIDc, TB), TB, 0, stream>>>(hu, h2, perm2, K2Pc, K1Pc);
  k_prep<<<grid((long)Bc * K1Pc * HIDc, TB), TB, 0, stream>>>(hu, HIDc, Wu0, dis1, sS, (
      _Float16*)nullptr, K1Pc);
  k_spmm_small<<<grid((long)Bc * K1Pc * HIDc, TB), TB, 0, stream>>>(A1, sS, dis1, bu0, g1, 1, K1Pc);

  k_copy<<<grid((long)Bc * Nc * HIDc, TB), TB, 0, stream>>>(hN, h0, (long)Bc * Nc * HIDc);
  k_scatter_add<<<grid((long)Bc * K1Pc * HIDc, TB), TB, 0, stream>>>(hN, g1, perm1, K1Pc, Nc);
  k_prep<<<grid((long)Bc * Nc * HIDc, TB), TB, 0, stream>>>(hN, HIDc, Wu1, dis0, s32, s16T, Nc);
  k_spmm_wmma<<<Bc * (Nc / 16), 32, 0, stream>>>(A8, s32, s16T, dis0, bu1, hfin, 0);

  k_bn_partial<<<64, 256, 0, stream>>>(hfin, part);
  k_bn_reduce<<<1, 64, 0, stream>>>(part, stats, 64);
  k_mlp<<<grid((long)Bc * Nc, TB), TB, 0, stream>>>(hfin, stats, gmm, bet,
                                                    W1, b1, W2, b2, W3, b3, out);
}
